// GRU_43215960932975
// MI455X (gfx1250) — compile-verified, matrix operand order checked by reading, not run on hardware
//
#include <hip/hip_runtime.h>
#include <math.h>

// ---------------------------------------------------------------------------
// GRU forward for MI455X (gfx1250): bf16 WMMA GEMMs (software-pipelined) +
// register-resident recurrence with LDS-resident recurrent weights (320KB WGP
// LDS), grid-wide sync, double-buffered WMMA-packed hidden state.
// ---------------------------------------------------------------------------

#define B_  32
#define T_  512
#define D_  1024
#define H_  1024
#define N3H (3 * H_)             // 3072
#define MT_GI   (B_ * T_ / 16)   // 1024 M-tiles for input projection
#define KT_D    (D_ / 32)        // 32 K-tiles (K step = 32 for bf16 WMMA)
#define KT_H    (H_ / 32)        // 32
#define NT_3H   (N3H / 16)       // 192 N-tiles
#define JT_H    (H_ / 16)        // 64 column tiles of the hidden state

typedef __attribute__((ext_vector_type(16))) __bf16 v16bf;
typedef __attribute__((ext_vector_type(8)))  __bf16 v8bf;
typedef __attribute__((ext_vector_type(8)))  float  v8f;

__device__ __forceinline__ v8f wmma_bf16(v16bf a, v16bf b, v8f c) {
  // v_wmma_f32_16x16x32_bf16 : D = A(16x32) * B(32x16) + C(16x16 f32)
  return __builtin_amdgcn_wmma_f32_16x16x32_bf16(
      /*neg_a=*/false, a, /*neg_b=*/false, b,
      /*c_mod=*/(short)0, c, /*reuse_a=*/false, /*reuse_b=*/false);
}

// ---- workspace layout (bytes) ---------------------------------------------
#define OFF_CNT 0                         // grid-sync counter
#define OFF_HA  4096                      // 2 x (2 mtiles x 32 ktiles x 512 bf16) = 131072 B
#define OFF_XA  262144                    // x packed A-tiles: 1024*32*512*2 = 32 MB
#define OFF_WIH (OFF_XA  + (size_t)MT_GI * KT_D * 512 * 2)   // 6 MB
#define OFF_WHH (OFF_WIH + (size_t)NT_3H * KT_D * 512 * 2)   // 6 MB
#define OFF_GI  (OFF_WHH + (size_t)NT_3H * KT_H * 512 * 2)   // gi tiles: 96 MB

// Recurrence dynamic LDS: 6KB gate-exchange + 192KB staged w_hh fragments
#define LDS_GH_BYTES  (6 * 256 * 4)
#define LDS_B_BYTES   (6 * KT_H * 512 * 2)
#define LDS_REC_BYTES (LDS_GH_BYTES + LDS_B_BYTES)   // 202752 < 320KB/WGP

// ---------------------------------------------------------------------------
__global__ void gru_init(unsigned* ws) {
  unsigned i = blockIdx.x * blockDim.x + threadIdx.x;
  const unsigned n = (OFF_HA + 131072) / 4;   // counter region + hA double buffer
  if (i < n) ws[i] = 0u;
}

// ---------------------------------------------------------------------------
// pack an [N x K] fp32 matrix into bf16 WMMA 16-bit fragment tiles (16x32).
//   lane = row(0..15) + 16*khalf ; elem e: v=e>>1, sub=e&1
//   K    = (v<4 ? 0:16) + khalf*8 + (v&3)*2 + sub          (ISA 7.12.2)
// ---------------------------------------------------------------------------
__global__ void gru_pack_frag(const float* __restrict__ src,
                              __bf16* __restrict__ dst,
                              int Ktiles, int Kdim) {
  int tile = blockIdx.x;                 // tile = nt * Ktiles + kt
  int nt = tile / Ktiles, kt = tile % Ktiles;
  for (int i = threadIdx.x; i < 512; i += 256) {
    int lane = i >> 4, e = i & 15;
    int rowL = lane & 15, khalf = lane >> 4;
    int v = e >> 1, sub = e & 1;
    int K = ((v < 4) ? 0 : 16) + khalf * 8 + (v & 3) * 2 + sub;
    dst[(size_t)tile * 512 + i] =
        (__bf16)src[(size_t)(nt * 16 + rowL) * Kdim + kt * 32 + K];
  }
}

// ---------------------------------------------------------------------------
// gi = x @ w_ih^T + b_ih, bf16 output in WMMA-C tile order.
// 1 M-tile x 4 N-tiles per wave; register double-buffered K loop so each
// s_wait_loadcnt covers loads issued one full WMMA group earlier.
// ---------------------------------------------------------------------------
__global__ __launch_bounds__(256)
void gru_gi_gemm(const __bf16* __restrict__ xA, const __bf16* __restrict__ wB,
                 const float* __restrict__ b_ih, __bf16* __restrict__ gi) {
  const int w = threadIdx.x >> 5, lane = threadIdx.x & 31;
  const int job = blockIdx.x * 8 + w;          // 1024 * 48 jobs
  const int mt = job / (NT_3H / 4);
  const int nt0 = (job % (NT_3H / 4)) * 4;

  const __bf16* aP = xA + (size_t)mt * KT_D * 512 + lane * 16;
  const __bf16* bP = wB + (size_t)nt0 * KT_D * 512 + lane * 16;

  v8f acc[4] = {v8f{}, v8f{}, v8f{}, v8f{}};
  v16bf a0 = *(const v16bf*)(aP);
  v16bf b0[4];
#pragma unroll
  for (int jj = 0; jj < 4; ++jj)
    b0[jj] = *(const v16bf*)(bP + (size_t)jj * KT_D * 512);

#pragma unroll 4
  for (int kt = 0; kt < KT_D; ++kt) {
    int ktn = (kt + 1) & (KT_D - 1);           // wrap: last iter reloads tile 0
    if (kt + 8 < KT_D)
      __builtin_prefetch((const void*)(aP + (kt + 8) * 512), 0, 1);
    v16bf a1 = *(const v16bf*)(aP + ktn * 512);
    v16bf b1[4];
#pragma unroll
    for (int jj = 0; jj < 4; ++jj)
      b1[jj] = *(const v16bf*)(bP + (size_t)jj * KT_D * 512 + ktn * 512);
#pragma unroll
    for (int jj = 0; jj < 4; ++jj)
      acc[jj] = wmma_bf16(a0, b0[jj], acc[jj]);
    a0 = a1;
#pragma unroll
    for (int jj = 0; jj < 4; ++jj) b0[jj] = b1[jj];
  }

  const int colL = lane & 15;
#pragma unroll
  for (int jj = 0; jj < 4; ++jj) {
    int nt = nt0 + jj;
    float bias = b_ih[nt * 16 + colL];
    v8bf o;
#pragma unroll
    for (int v = 0; v < 8; ++v) o[v] = (__bf16)(acc[jj][v] + bias);
    *(v8bf*)(gi + ((size_t)mt * NT_3H + nt) * 256 + lane * 8) = o;  // b128
  }
}

// ---------------------------------------------------------------------------
// Recurrence: 64 blocks (one per 16-col tile j of H) x 6 waves.
// Wave w: gate g=w>>1 (r,z,n), m-tile m=w&1 -> one 16x16 gh tile, K=1024.
// w_hh fragments staged to 192KB of LDS once (CDNA5: 320KB LDS/WGP); per-step
// B operands are ds_loads, A (fresh hidden state) is prefetched + pipelined.
// ---------------------------------------------------------------------------
__global__ __launch_bounds__(192)
void gru_recurrence(__bf16* __restrict__ hA, const __bf16* __restrict__ whhB,
                    const __bf16* __restrict__ gi, const float* __restrict__ b_hh,
                    const int* __restrict__ seq_len, float* __restrict__ out,
                    unsigned* __restrict__ cnt) {
  extern __shared__ char smem[];
  float (*gh)[256] = (float(*)[256])smem;                 // 6 x 256 f32
  __bf16* ldsB = (__bf16*)(smem + LDS_GH_BYTES);          // 6 x 32 x 512 bf16

  const int tid = threadIdx.x;
  const int w = tid >> 5, lane = tid & 31;
  const int j = blockIdx.x;            // H column tile 0..63
  const int g = w >> 1, m = w & 1;
  const int ntg = g * JT_H + j;        // N-tile within 3H
  const int colL = lane & 15;
  const int col = j * 16 + colL;
  const unsigned nb = gridDim.x;

  // ---- stage this wave's w_hh fragment slice into LDS (once) ----
  {
    v8bf* dstv = (v8bf*)(ldsB + (size_t)w * KT_H * 512);
    const v8bf* srcv = (const v8bf*)(whhB + (size_t)ntg * KT_H * 512);
    for (int i = lane; i < KT_H * 512 / 8; i += 32) dstv[i] = srcv[i];
  }

  float hreg[8];
  int   sl[8];
  float bhr = 0.f, bhz = 0.f, bhn = 0.f;
  if (w < 2) {
#pragma unroll
    for (int v = 0; v < 8; ++v) {
      hreg[v] = 0.f;
      sl[v] = seq_len[m * 16 + ((lane >> 4) * 8) + v];
    }
    bhr = b_hh[col]; bhz = b_hh[H_ + col]; bhn = b_hh[2 * H_ + col];
  }
  __syncthreads();   // LDS B staging complete

  const __bf16* ldsBme = ldsB + (size_t)w * KT_H * 512 + lane * 16;

  for (int t = 0; t < T_; ++t) {
    const __bf16* hRead  = hA + (size_t)(t & 1) * 32768;
    __bf16*       hWrite = hA + (size_t)((t + 1) & 1) * 32768;
    const __bf16* aP = hRead + (size_t)(m * KT_H) * 512 + lane * 16;

    // warm WGP caches with the freshly written hidden-state fragments
#pragma unroll
    for (int kt = 0; kt < KT_H; kt += 4)
      __builtin_prefetch((const void*)(aP + kt * 512), 0, 3);

    // gh tile: 32 chained bf16 WMMAs; A pipelined (global), B from LDS
    v8f acc = {};
    v16bf a0 = *(const v16bf*)(aP);
#pragma unroll 4
    for (int kt = 0; kt < KT_H; ++kt) {
      int ktn = (kt + 1) & (KT_H - 1);
      v16bf a1 = *(const v16bf*)(aP + ktn * 512);
      v16bf b  = *(const v16bf*)(ldsBme + kt * 512);   // ds_load_b128 x2
      acc = wmma_bf16(a0, b, acc);
      a0 = a1;
    }
#pragma unroll
    for (int v = 0; v < 8; ++v) gh[w][lane * 8 + v] = acc[v];
    __syncthreads();

    if (w < 2) {                        // pointwise gate math for m-tile m
#pragma unroll
      for (int v = 0; v < 8; ++v) {
        int r_ = (lane >> 4) * 8 + v;   // row within m-tile
        int b  = m * 16 + r_;           // batch index
        int gimt = (b * T_ + t) >> 4;
        int rlo = t & 15;
        int eoff = (colL + ((rlo >> 3) << 4)) * 8 + (rlo & 7);
        float ir = (float)gi[((size_t)gimt * NT_3H + 0 * JT_H + j) * 256 + eoff];
        float iz = (float)gi[((size_t)gimt * NT_3H + 1 * JT_H + j) * 256 + eoff];
        float in = (float)gi[((size_t)gimt * NT_3H + 2 * JT_H + j) * 256 + eoff];
        int lidx = (colL + ((r_ >> 3) << 4)) * 8 + (r_ & 7);
        float hr = gh[0 * 2 + m][lidx] + bhr;
        float hz = gh[1 * 2 + m][lidx] + bhz;
        float hn = gh[2 * 2 + m][lidx] + bhn;
        float rg = 1.f / (1.f + __expf(-(ir + hr)));
        float zg = 1.f / (1.f + __expf(-(iz + hz)));
        float ng = tanhf(in + rg * hn);
        float hold = hreg[v];
        float hnew = (1.f - zg) * ng + zg * hold;
        bool valid = (t < sl[v]);
        float hnext = valid ? hnew : hold;
        hreg[v] = hnext;
        out[((size_t)b * T_ + t) * H_ + col] = valid ? hnew : 0.f;
        // repack hnext into bf16 A-fragment layout for next step's WMMA
        int ktp = col >> 5, Kl = col & 31;
        int khalf = (Kl >> 3) & 1;
        int top = (Kl >= 16) ? 16 : 0;
        int within = Kl - top - khalf * 8;
        int e = ((top ? 4 : 0) + (within >> 1)) * 2 + (within & 1);
        int lane_p = (b & 15) + (khalf << 4);
        hWrite[(size_t)(m * KT_H + ktp) * 512 + lane_p * 16 + e] = (__bf16)hnext;
      }
    }

    // grid-wide barrier: all blocks' hWrite visible before next step reads it
    __threadfence();
    __syncthreads();
    if (tid == 0) {
      atomicAdd(cnt, 1u);
      unsigned target = nb * (unsigned)(t + 1);
      while (*(volatile unsigned*)cnt < target) __builtin_amdgcn_s_sleep(1);
    }
    __syncthreads();
  }

  if (w < 2) {                          // h_n = last valid hidden (held in regs)
#pragma unroll
    for (int v = 0; v < 8; ++v) {
      int b = m * 16 + ((lane >> 4) * 8) + v;
      out[(size_t)B_ * T_ * H_ + (size_t)b * H_ + col] = hreg[v];
    }
  }
}

// ---------------------------------------------------------------------------
extern "C" void kernel_launch(void* const* d_in, const int* in_sizes, int n_in,
                              void* d_out, int out_size, void* d_ws, size_t ws_size,
                              hipStream_t stream) {
  const float* x       = (const float*)d_in[0];
  const int*   seq_len = (const int*)  d_in[1];
  const float* w_ih    = (const float*)d_in[2];
  const float* w_hh    = (const float*)d_in[3];
  const float* b_ih    = (const float*)d_in[4];
  const float* b_hh    = (const float*)d_in[5];
  float* out = (float*)d_out;
  char* ws = (char*)d_ws;

  __bf16* hA   = (__bf16*)(ws + OFF_HA);
  __bf16* xA   = (__bf16*)(ws + OFF_XA);
  __bf16* wihB = (__bf16*)(ws + OFF_WIH);
  __bf16* whhB = (__bf16*)(ws + OFF_WHH);
  __bf16* gi   = (__bf16*)(ws + OFF_GI);
  unsigned* cnt = (unsigned*)(ws + OFF_CNT);

  // allow 192KB+ dynamic LDS for the recurrence kernel (320KB/WGP on CDNA5)
  (void)hipFuncSetAttribute((const void*)gru_recurrence,
                            hipFuncAttributeMaxDynamicSharedMemorySize,
                            LDS_REC_BYTES);

  // 1) zero counter + hA double buffer (h0 = 0)
  gru_init<<<((OFF_HA + 131072) / 4 + 255) / 256, 256, 0, stream>>>((unsigned*)ws);
  // 2) pack x, w_ih, w_hh into bf16 WMMA fragment tiles
  gru_pack_frag<<<MT_GI * KT_D, 256, 0, stream>>>(x, xA, KT_D, D_);
  gru_pack_frag<<<NT_3H * KT_D, 256, 0, stream>>>(w_ih, wihB, KT_D, D_);
  gru_pack_frag<<<NT_3H * KT_H, 256, 0, stream>>>(w_hh, whhB, KT_H, H_);
  // 3) input projection GEMM (bf16 WMMA, fp32 accumulate)
  gru_gi_gemm<<<(MT_GI * (NT_3H / 4)) / 8, 256, 0, stream>>>(xA, wihB, b_ih, gi);
  // 4) sequential recurrence: 64 blocks x 6 waves, grid-synced per step
  gru_recurrence<<<JT_H, 192, LDS_REC_BYTES, stream>>>(hA, whhB, gi, b_hh,
                                                       seq_len, out, cnt);
}
